// Qwen3Attention_55113020342361
// MI455X (gfx1250) — matrix operand order repaired and checked by reading, not compiled, not run on hardware
//
#include <hip/hip_runtime.h>

// ---------------------------------------------------------------------------
// Types
// ---------------------------------------------------------------------------
typedef __attribute__((ext_vector_type(16))) __bf16 v16bf;
typedef __attribute__((ext_vector_type(8)))  float  v8f;
typedef __attribute__((ext_vector_type(4)))  unsigned int u32x4;
typedef __attribute__((ext_vector_type(4)))  unsigned int tdg0_t;
typedef __attribute__((ext_vector_type(8)))  int          tdg1_t;
typedef __attribute__((ext_vector_type(4)))  int          tdg2_t;
typedef __attribute__((ext_vector_type(8)))  int          tdg4_t;

union FragBF {
    u32x4 u[2];
    v16bf v;
    unsigned short s[16];
};

__device__ __forceinline__ unsigned short f2bf(float f) {
    __bf16 b = (__bf16)f;
    return __builtin_bit_cast(unsigned short, b);
}

// Problem constants (Qwen3: B=16, S=512, D_MODEL=4096, H=32, KV=8, HD=128)
#define BB      16
#define SS      512
#define DMODEL  4096
#define NH      32
#define NKV     8
#define HD      128
#define MTOK    (BB * SS)          // 8192 tokens

// ---------------------------------------------------------------------------
// Tensor Data Mover: issue a 2D tile load (bf16 elements) into LDS.
// D# per CDNA5 ISA.  All inputs must be wave-uniform.
//   group0: [1:0]=count=1, [63:32]=lds byte addr, [120:64]=global addr,
//           [127:126]=type=2
//   group1: [17:16]=data_size(1=2B), [79:48]=tensor_dim0, [111:80]=tensor_dim1,
//           [127:112]=tile_dim0, [143:128]=tile_dim1, [207:160]=dim0_stride
// This toolchain exposes the 6-arg builtin (clang-23 / therock headers).
// ---------------------------------------------------------------------------
__device__ __forceinline__ void tdm_load_2d_bf16(unsigned lds_addr,
                                                 const void* gptr,
                                                 unsigned tile_d0,
                                                 unsigned tile_d1,
                                                 unsigned tensor_d0,
                                                 unsigned tensor_d1,
                                                 unsigned long long stride0) {
    unsigned long long ga = (unsigned long long)gptr;
    tdg0_t g0;
    g0[0] = 1u;                                     // count=1, user descriptor
    g0[1] = lds_addr;                               // LDS byte address
    g0[2] = (unsigned)(ga & 0xffffffffu);           // global_addr[31:0]
    g0[3] = (unsigned)((ga >> 32) & 0x01ffffffu)    // global_addr[56:32]
          | 0x80000000u;                            // type=2 ("image")
    tdg1_t g1;
    g1[0] = (int)(1u << 16);                        // data_size = 1 -> 2 bytes
    g1[1] = (int)((tensor_d0 & 0xffffu) << 16);     // tensor_dim0[15:0]
    g1[2] = (int)((tensor_d0 >> 16) & 0xffffu)      // tensor_dim0[31:16]
          | (int)((tensor_d1 & 0xffffu) << 16);     // tensor_dim1[15:0]
    g1[3] = (int)((tensor_d1 >> 16) & 0xffffu)      // tensor_dim1[31:16]
          | (int)((tile_d0 & 0xffffu) << 16);       // tile_dim0
    g1[4] = (int)(tile_d1 & 0xffffu);               // tile_dim1 (tile_dim2=0)
    g1[5] = (int)(unsigned)(stride0 & 0xffffffffull);        // dim0_stride lo32
    g1[6] = (int)(unsigned)((stride0 >> 32) & 0xffffull);    // dim0_stride hi16
    g1[7] = 0;                                      // dim1_stride hi = 0
    tdg2_t z4 = {0, 0, 0, 0};
    tdg4_t z8 = {0, 0, 0, 0, 0, 0, 0, 0};
    __builtin_amdgcn_tensor_load_to_lds(g0, g1, z4, z4, z8, 0);
}

// ---------------------------------------------------------------------------
// fp32 -> bf16 convert (grid-stride)
// ---------------------------------------------------------------------------
__global__ void cvt_f32_bf16(const float* __restrict__ in,
                             unsigned short* __restrict__ out, long n) {
    long i = (long)blockIdx.x * blockDim.x + threadIdx.x;
    long stride = (long)gridDim.x * blockDim.x;
    for (; i < n; i += stride) out[i] = f2bf(in[i]);
}

// ---------------------------------------------------------------------------
// bf16 GEMM:  C[M,N] (f32) = A[M,K] (bf16, row major) * W[K,N] (bf16, row major)
// Block tile 128x128, K-step 32, 256 threads = 8 waves, wave tile 64x32.
// A tile staged by the Tensor Data Mover (wave 0 issues, TENSORcnt sync);
// B tile staged transposed by all threads.  W rows prefetched one K-step ahead.
// ---------------------------------------------------------------------------
#define BSTR 40   // padded k-stride (halfs) of transposed B tile: 80B, 16B aligned

__launch_bounds__(256)
__global__ void gemm_bf16(const unsigned short* __restrict__ A,
                          const unsigned short* __restrict__ W,
                          float* __restrict__ C,
                          int M, int N, int K) {
    __shared__ __align__(16) unsigned short As[128 * 32];
    __shared__ __align__(16) unsigned short Bs[128 * BSTR];

    const int tid   = threadIdx.x;
    const int lane  = tid & 31;
    const int wave  = tid >> 5;
    const int lhalf = lane >> 4;
    const int l15   = lane & 15;
    const int wm    = (wave >> 2) * 64;   // wave M offset in block tile
    const int wn    = (wave & 3) * 32;    // wave N offset in block tile
    const int bm    = blockIdx.y * 128;
    const int bn    = blockIdx.x * 128;

    const unsigned as_lds = (unsigned)(unsigned long long)(void*)As;

    v8f acc[4][2] = {};

    for (int kb = 0; kb < K; kb += 32) {
        __syncthreads();
        // --- stage A tile [128 rows][32 k] via TDM (wave 0 issues the DMA) ---
        if (wave == 0) {
            tdm_load_2d_bf16(as_lds, A + (long)bm * K + kb,
                             /*tile*/ 32, 128, /*tensor*/ (unsigned)K,
                             (unsigned)M, (unsigned long long)K);
        }
        // --- stage B tile transposed: Bs[n][k], padded stride ---
        {
            int k = tid >> 3, nch = (tid & 7) * 16;
            FragBF t;
            const u32x4* src = (const u32x4*)(W + (long)(kb + k) * N + bn + nch);
            t.u[0] = src[0];
            t.u[1] = src[1];
#pragma unroll
            for (int i = 0; i < 16; i++) Bs[(nch + i) * BSTR + k] = t.s[i];
            if (kb + 32 < K)   // pull next K-step of W into GL2 while WMMAs run
                __builtin_prefetch(W + (long)(kb + 32 + k) * N + bn + nch, 0, 1);
        }
        if (wave == 0) __builtin_amdgcn_s_wait_tensorcnt(0);
        __syncthreads();

        // --- fragment loads + 8 WMMAs per wave ---
        FragBF a[4], b[2];
#pragma unroll
        for (int mt = 0; mt < 4; mt++) {
            const unsigned short* base = As + (wm + mt * 16 + l15) * 32 + lhalf * 8;
            a[mt].u[0] = *(const u32x4*)(base);        // K = lhalf*8 + 0..7
            a[mt].u[1] = *(const u32x4*)(base + 16);   // K = lhalf*8 + 16..23
        }
#pragma unroll
        for (int nt = 0; nt < 2; nt++) {
            const unsigned short* base = Bs + (wn + nt * 16 + l15) * BSTR + lhalf * 8;
            b[nt].u[0] = *(const u32x4*)(base);        // K = lhalf*8 + 0..7
            b[nt].u[1] = *(const u32x4*)(base + 16);   // K = lhalf*8 + 16..23
        }
#pragma unroll
        for (int mt = 0; mt < 4; mt++)
#pragma unroll
            for (int nt = 0; nt < 2; nt++)
                acc[mt][nt] = __builtin_amdgcn_wmma_f32_16x16x32_bf16(
                    false, a[mt].v, false, b[nt].v, (short)0, acc[mt][nt],
                    false, false);
    }

    // --- store C: one base pointer per tile, constant row offsets (r*N) ---
#pragma unroll
    for (int mt = 0; mt < 4; mt++)
#pragma unroll
        for (int nt = 0; nt < 2; nt++) {
            float* cp = C + (long)(bm + wm + mt * 16 + 8 * lhalf) * N
                          + bn + wn + nt * 16 + l15;
#pragma unroll
            for (int r = 0; r < 8; r++) cp[(long)r * N] = acc[mt][nt][r];
        }
}

// ---------------------------------------------------------------------------
// Per-head RMSNorm + RoPE.  One wave per (token, slot):
//   slot 0..31  -> q head (norm + rope)
//   slot 32..39 -> k head (norm + rope)
//   slot 40..47 -> v head (convert only)
// Lane owns dims {l, l+32, l+64, l+96} so rotate-half pairs stay in-lane.
// ---------------------------------------------------------------------------
__launch_bounds__(256)
__global__ void normrope(const float* __restrict__ qf,
                         const float* __restrict__ kf,
                         const float* __restrict__ vf,
                         const int* __restrict__ pos_ids,
                         const float* __restrict__ qw,
                         const float* __restrict__ kw,
                         unsigned short* __restrict__ qb,
                         unsigned short* __restrict__ kb,
                         unsigned short* __restrict__ vb) {
    const int lane = threadIdx.x & 31;
    const int wave = threadIdx.x >> 5;
    long gid   = (long)blockIdx.x * 8 + wave;
    long token = gid / 48;
    int  slot  = (int)(gid % 48);

    const float* src; unsigned short* dst; const float* w;
    bool dorope = true;
    if (slot < 32) {
        src = qf + token * (NH * HD)  + slot * HD;
        dst = qb + token * (NH * HD)  + slot * HD;
        w = qw;
    } else if (slot < 40) {
        src = kf + token * (NKV * HD) + (slot - 32) * HD;
        dst = kb + token * (NKV * HD) + (slot - 32) * HD;
        w = kw;
    } else {
        src = vf + token * (NKV * HD) + (slot - 40) * HD;
        dst = vb + token * (NKV * HD) + (slot - 40) * HD;
        w = nullptr; dorope = false;
    }

    float x0 = src[lane], x1 = src[lane + 32], x2 = src[lane + 64], x3 = src[lane + 96];
    if (!dorope) {
        dst[lane]      = f2bf(x0);
        dst[lane + 32] = f2bf(x1);
        dst[lane + 64] = f2bf(x2);
        dst[lane + 96] = f2bf(x3);
        return;
    }

    float ss = x0 * x0 + x1 * x1 + x2 * x2 + x3 * x3;
#pragma unroll
    for (int m = 16; m >= 1; m >>= 1) ss += __shfl_xor(ss, m, 32);
    float r = rsqrtf(ss * (1.0f / 128.0f) + 1e-6f);
    x0 = x0 * r * w[lane];
    x1 = x1 * r * w[lane + 32];
    x2 = x2 * r * w[lane + 64];
    x3 = x3 * r * w[lane + 96];

    float p = (float)pos_ids[token];   // position_ids is arange(S) broadcast over B
    const float LOG_THETA = 13.815510557964274f;  // ln(1e6)
    float f0 = p * __expf(-((float)(2 * lane)        * (1.0f / 128.0f)) * LOG_THETA);
    float f1 = p * __expf(-((float)(2 * (lane + 32)) * (1.0f / 128.0f)) * LOG_THETA);
    float s0, c0, s1, c1;
    __sincosf(f0, &s0, &c0);
    __sincosf(f1, &s1, &c1);

    dst[lane]      = f2bf(x0 * c0 - x2 * s0);
    dst[lane + 32] = f2bf(x1 * c1 - x3 * s1);
    dst[lane + 64] = f2bf(x2 * c0 + x0 * s0);
    dst[lane + 96] = f2bf(x3 * c1 + x1 * s1);
}

// ---------------------------------------------------------------------------
// Fused causal flash attention with GQA (groups = 4).
// grid = (S/64, NH, B), block = 128 (4 waves).  Wave w owns q rows
// [qsuper*64 + w*16, +16).  Per 32-key block: K tile staged by the TDM
// (wave 0, TENSORcnt), V^T staged manually, QK^T via WMMA, online softmax
// (width-16 shuffles), P re-laid out through per-wave LDS scratch into
// A-fragment format, PV via WMMA.
// ---------------------------------------------------------------------------
__launch_bounds__(128)
__global__ void flash_attn(const unsigned short* __restrict__ qb,
                           const unsigned short* __restrict__ kbuf,
                           const unsigned short* __restrict__ vbuf,
                           unsigned short* __restrict__ ob) {
    __shared__ __align__(16) unsigned short Kt[32 * 128];       // [key][d]
    __shared__ __align__(16) unsigned short Vt[128 * 32];       // [d][key]
    __shared__ __align__(16) unsigned short Pw[4][16 * 32];     // per-wave P scratch

    const int tid    = threadIdx.x;
    const int lane   = tid & 31;
    const int wave   = tid >> 5;
    const int lhalf  = lane >> 4;
    const int l15    = lane & 15;
    const int qsuper = blockIdx.x;           // 0..7
    const int h      = blockIdx.y;           // 0..31
    const int b      = blockIdx.z;           // 0..15
    const int kvh    = h >> 2;               // GQA group
    const int qbase  = qsuper * 64 + wave * 16;
    const float scale = 0.08838834764831845f;   // 1/sqrt(128)

    const unsigned kt_lds = (unsigned)(unsigned long long)(void*)Kt;

    // Q A-fragments (4 chunks of 16x32), kept in registers for the whole loop
    FragBF aq[4];
    {
        long qrow = ((long)(b * SS + qbase + l15) * NH + h) * HD;
#pragma unroll
        for (int dc = 0; dc < 4; dc++) {
            const unsigned short* base = qb + qrow + dc * 32 + lhalf * 8;
            aq[dc].u[0] = *(const u32x4*)(base);
            aq[dc].u[1] = *(const u32x4*)(base + 16);
        }
    }

    v8f accO[8] = {};
    float mrow[8], lrow[8];
#pragma unroll
    for (int r = 0; r < 8; r++) { mrow[r] = -3.0e38f; lrow[r] = 0.0f; }

    const int kend = qsuper * 64 + 64;
    for (int kb0 = 0; kb0 < kend; kb0 += 32) {
        __syncthreads();
        // --- K tile [32 keys][128 d] via TDM: rows strided NKV*HD elements ---
        if (wave == 0) {
            const unsigned short* kg =
                kbuf + ((long)(b * SS + kb0) * NKV + kvh) * HD;
            tdm_load_2d_bf16(kt_lds, kg, /*tile*/ HD, 32,
                             /*tensor*/ HD, SS,
                             (unsigned long long)(NKV * HD));
        }
        // --- stage V transposed: Vt[d][key] ---
        {
            int key = tid >> 2, dch = (tid & 3) * 32;
            long srow = ((long)(b * SS + kb0 + key) * NKV + kvh) * HD + dch;
            FragBF t0, t1;
            const u32x4* vs = (const u32x4*)(vbuf + srow);
            t0.u[0] = vs[0]; t0.u[1] = vs[1];
            t1.u[0] = vs[2]; t1.u[1] = vs[3];
#pragma unroll
            for (int i = 0; i < 16; i++) Vt[(dch + i) * 32 + key]      = t0.s[i];
#pragma unroll
            for (int i = 0; i < 16; i++) Vt[(dch + 16 + i) * 32 + key] = t1.s[i];
        }
        if (wave == 0) __builtin_amdgcn_s_wait_tensorcnt(0);
        __syncthreads();

        // --- scores: two 16-key subtiles, 4 d-chunks each ---
        v8f sc[2] = {};
#pragma unroll
        for (int j = 0; j < 2; j++)
#pragma unroll
            for (int dc = 0; dc < 4; dc++) {
                FragBF bk;
                const unsigned short* base =
                    Kt + (j * 16 + l15) * 128 + dc * 32 + lhalf * 8;
                bk.u[0] = *(const u32x4*)(base);
                bk.u[1] = *(const u32x4*)(base + 16);
                sc[j] = __builtin_amdgcn_wmma_f32_16x16x32_bf16(
                    false, aq[dc].v, false, bk.v, (short)0, sc[j], false, false);
            }

        // --- causal mask + online softmax (per-row over 16-lane halves) ---
        float scl[8];
#pragma unroll
        for (int r = 0; r < 8; r++) {
            int qrow = qbase + r + 8 * lhalf;
            float s0 = sc[0][r] * scale;
            float s1 = sc[1][r] * scale;
            if (kb0 + l15 > qrow)      s0 = -3.0e38f;
            if (kb0 + 16 + l15 > qrow) s1 = -3.0e38f;
            float mx = fmaxf(s0, s1);
            mx = fmaxf(mx, __shfl_xor(mx, 1, 16));
            mx = fmaxf(mx, __shfl_xor(mx, 2, 16));
            mx = fmaxf(mx, __shfl_xor(mx, 4, 16));
            mx = fmaxf(mx, __shfl_xor(mx, 8, 16));
            float mnew = fmaxf(mrow[r], mx);
            float p0 = __expf(s0 - mnew), p1 = __expf(s1 - mnew);
            float ps = p0 + p1;
            ps += __shfl_xor(ps, 1, 16);
            ps += __shfl_xor(ps, 2, 16);
            ps += __shfl_xor(ps, 4, 16);
            ps += __shfl_xor(ps, 8, 16);
            float corr = __expf(mrow[r] - mnew);
            lrow[r] = lrow[r] * corr + ps;
            mrow[r] = mnew;
            scl[r]  = corr;
            int prow = r + 8 * lhalf;
            Pw[wave][prow * 32 + l15]      = f2bf(p0);
            Pw[wave][prow * 32 + 16 + l15] = f2bf(p1);
        }

        // --- rescale running output, then PV ---
#pragma unroll
        for (int dt = 0; dt < 8; dt++)
#pragma unroll
            for (int r = 0; r < 8; r++) accO[dt][r] *= scl[r];

        FragBF pa;   // P as A-fragment (16 q-rows x 32 keys)
        {
            const unsigned short* base = &Pw[wave][l15 * 32 + lhalf * 8];
            pa.u[0] = *(const u32x4*)(base);
            pa.u[1] = *(const u32x4*)(base + 16);
        }
#pragma unroll
        for (int dt = 0; dt < 8; dt++) {
            FragBF bv;
            const unsigned short* base = Vt + (dt * 16 + l15) * 32 + lhalf * 8;
            bv.u[0] = *(const u32x4*)(base);
            bv.u[1] = *(const u32x4*)(base + 16);
            accO[dt] = __builtin_amdgcn_wmma_f32_16x16x32_bf16(
                false, pa.v, false, bv.v, (short)0, accO[dt], false, false);
        }
    }

    // --- epilogue: normalize, bf16 store [B,S,H,D] with constant row offsets ---
#pragma unroll
    for (int dt = 0; dt < 8; dt++) {
        unsigned short* op = ob
            + ((long)(b * SS + qbase + 8 * lhalf) * NH + h) * HD + dt * 16 + l15;
#pragma unroll
        for (int r = 0; r < 8; r++)
            op[(long)r * NH * HD] = f2bf(accO[dt][r] / lrow[r]);
    }
}

// ---------------------------------------------------------------------------
// Host launcher
// ---------------------------------------------------------------------------
extern "C" void kernel_launch(void* const* d_in, const int* in_sizes, int n_in,
                              void* d_out, int out_size, void* d_ws, size_t ws_size,
                              hipStream_t stream) {
    const float* hidden = (const float*)d_in[0];
    const int*   pos    = (const int*)d_in[1];
    const float* Wq     = (const float*)d_in[2];
    const float* Wk     = (const float*)d_in[3];
    const float* Wv     = (const float*)d_in[4];
    const float* Wo     = (const float*)d_in[5];
    const float* qw     = (const float*)d_in[6];
    const float* kw     = (const float*)d_in[7];

    char* ws = (char*)d_ws;
    size_t off = 0;
    auto alloc = [&](size_t bytes) {
        char* p = ws + off;
        off += (bytes + 255) & ~(size_t)255;
        return p;
    };
    unsigned short* hsb   = (unsigned short*)alloc((size_t)MTOK * DMODEL * 2);
    unsigned short* wqb   = (unsigned short*)alloc((size_t)DMODEL * (NH * HD) * 2);
    unsigned short* wkb   = (unsigned short*)alloc((size_t)DMODEL * (NKV * HD) * 2);
    unsigned short* wvb   = (unsigned short*)alloc((size_t)DMODEL * (NKV * HD) * 2);
    unsigned short* wob   = (unsigned short*)alloc((size_t)(NH * HD) * DMODEL * 2);
    float*          qf    = (float*)alloc((size_t)MTOK * (NH * HD) * 4);
    float*          kf    = (float*)alloc((size_t)MTOK * (NKV * HD) * 4);
    float*          vf    = (float*)alloc((size_t)MTOK * (NKV * HD) * 4);
    unsigned short* qbb   = (unsigned short*)alloc((size_t)MTOK * (NH * HD) * 2);
    unsigned short* kbb   = (unsigned short*)alloc((size_t)MTOK * (NKV * HD) * 2);
    unsigned short* vbb   = (unsigned short*)alloc((size_t)MTOK * (NKV * HD) * 2);
    unsigned short* attnb = (unsigned short*)alloc((size_t)MTOK * (NH * HD) * 2);

    // 1) convert fp32 -> bf16
    cvt_f32_bf16<<<8192, 256, 0, stream>>>(hidden, hsb, (long)MTOK * DMODEL);
    cvt_f32_bf16<<<8192, 256, 0, stream>>>(Wq, wqb, (long)DMODEL * NH * HD);
    cvt_f32_bf16<<<4096, 256, 0, stream>>>(Wk, wkb, (long)DMODEL * NKV * HD);
    cvt_f32_bf16<<<4096, 256, 0, stream>>>(Wv, wvb, (long)DMODEL * NKV * HD);
    cvt_f32_bf16<<<8192, 256, 0, stream>>>(Wo, wob, (long)NH * HD * DMODEL);

    // 2) QKV projections (bf16 WMMA GEMM, fp32 out)
    gemm_bf16<<<dim3((NH * HD) / 128, MTOK / 128), 256, 0, stream>>>(
        hsb, wqb, qf, MTOK, NH * HD, DMODEL);
    gemm_bf16<<<dim3((NKV * HD) / 128, MTOK / 128), 256, 0, stream>>>(
        hsb, wkb, kf, MTOK, NKV * HD, DMODEL);
    gemm_bf16<<<dim3((NKV * HD) / 128, MTOK / 128), 256, 0, stream>>>(
        hsb, wvb, vf, MTOK, NKV * HD, DMODEL);

    // 3) RMSNorm + RoPE -> bf16 Q/K/V
    normrope<<<(MTOK * 48) / 8, 256, 0, stream>>>(qf, kf, vf, pos, qw, kw,
                                                  qbb, kbb, vbb);

    // 4) fused causal flash attention (GQA)
    flash_attn<<<dim3(SS / 64, NH, BB), 128, 0, stream>>>(qbb, kbb, vbb, attnb);

    // 5) output projection -> fp32 d_out
    gemm_bf16<<<dim3(DMODEL / 128, MTOK / 128), 256, 0, stream>>>(
        attnb, wob, (float*)d_out, MTOK, DMODEL, NH * HD);
}
